// AfmoeMoE_75737453297753
// MI455X (gfx1250) — matrix operand order repaired and checked by reading, not compiled, not run on hardware
//
#include <hip/hip_runtime.h>
#include <hip/hip_fp16.h>
#include <math.h>

// Problem dims (match reference)
static constexpr int T = 1024;     // tokens
static constexpr int H = 1024;     // hidden
static constexpr int E = 32;       // experts
static constexpr int I = 512;      // moe intermediate
static constexpr int TOPK = 4;
static constexpr int NGRP = 4;     // expert groups
static constexpr int GSZ = E / NGRP;  // 8 experts per group
static constexpr int S = T * TOPK; // total routed slots = 4096
static constexpr float ROUTE_SCALE = 2.5f;

typedef __attribute__((ext_vector_type(16))) _Float16 v16h;
typedef __attribute__((ext_vector_type(8)))  _Float16 v8h;
typedef __attribute__((ext_vector_type(8)))  float    v8f;

// ---------------------------------------------------------------------------
// Small utility kernels
// ---------------------------------------------------------------------------
__global__ void f32_to_f16_kernel(const float* __restrict__ in,
                                  _Float16* __restrict__ out, int n) {
    int i = blockIdx.x * 256 + threadIdx.x;
    if (i < n) out[i] = (_Float16)in[i];
}

// Transpose fp32 [R x C] (C fastest) -> f16 [C x R], batched over blockIdx.z.
__global__ void transpose_f32_f16_kernel(const float* __restrict__ in,
                                         _Float16* __restrict__ out,
                                         int R, int C) {
    __shared__ float tile[32][33];
    const float* src = in + (size_t)blockIdx.z * R * C;
    _Float16* dst = out + (size_t)blockIdx.z * R * C;
    int c0 = blockIdx.x * 32, r0 = blockIdx.y * 32;
    for (int i = threadIdx.y; i < 32; i += 8)
        tile[i][threadIdx.x] = src[(size_t)(r0 + i) * C + c0 + threadIdx.x];
    __syncthreads();
    for (int i = threadIdx.y; i < 32; i += 8)
        dst[(size_t)(c0 + i) * R + r0 + threadIdx.x] = (_Float16)tile[threadIdx.x][i];
}

__global__ void zero_i32_kernel(int* p, int n) {
    int i = blockIdx.x * 256 + threadIdx.x;
    if (i < n) p[i] = 0;
}

// ---------------------------------------------------------------------------
// Router: one wave32 per token, lane == expert (E == 32). fp32 exact math.
// ---------------------------------------------------------------------------
__global__ void router_kernel(const float* __restrict__ x,
                              const float* __restrict__ gate_w,
                              const float* __restrict__ bias,
                              int* __restrict__ topk_id,
                              float* __restrict__ topk_w) {
    __shared__ float sb_s[8][32];   // biased sigmoid scores
    __shared__ float sc_s[8][32];   // raw sigmoid scores
    int wave = threadIdx.x >> 5, lane = threadIdx.x & 31;
    int t = blockIdx.x * 8 + wave;
    const float* xr = x + (size_t)t * H;
    const float* wr = gate_w + (size_t)lane * H;
    float acc = 0.f;
    for (int h = 0; h < H; ++h) acc += xr[h] * wr[h];
    float sc = 1.f / (1.f + expf(-acc));
    sc_s[wave][lane] = sc;
    sb_s[wave][lane] = sc + bias[lane];
    __syncthreads();
    if (lane == 0) {
        float* sb = sb_s[wave];
        float* scv = sc_s[wave];
        float gs[NGRP];
        for (int g = 0; g < NGRP; ++g) {
            float m1 = -1e30f, m2 = -1e30f;
            for (int j = 0; j < GSZ; ++j) {
                float v = sb[g * GSZ + j];
                if (v > m1) { m2 = m1; m1 = v; } else if (v > m2) { m2 = v; }
            }
            gs[g] = m1 + m2;
        }
        // top-2 groups (ties -> lower index, matches jax top_k)
        int g1 = 0;
        for (int g = 1; g < NGRP; ++g) if (gs[g] > gs[g1]) g1 = g;
        int g2 = -1;
        for (int g = 0; g < NGRP; ++g) {
            if (g == g1) continue;
            if (g2 < 0 || gs[g] > gs[g2]) g2 = g;
        }
        float masked[E];
        for (int e = 0; e < E; ++e) {
            int g = e / GSZ;
            masked[e] = (g == g1 || g == g2) ? sb[e] : -1e30f;
        }
        int ids[TOPK]; float wv[TOPK]; float wsum = 0.f;
        for (int k = 0; k < TOPK; ++k) {
            int best = 0; float bv = -3e30f;
            for (int e = 0; e < E; ++e)
                if (masked[e] > bv) { bv = masked[e]; best = e; }
            ids[k] = best;
            masked[best] = -2e30f;
            wv[k] = scv[best];
            wsum += wv[k];
        }
        float s = ROUTE_SCALE / wsum;
        for (int k = 0; k < TOPK; ++k) {
            topk_id[t * TOPK + k] = ids[k];
            topk_w[t * TOPK + k] = wv[k] * s;
        }
    }
}

// ---------------------------------------------------------------------------
// Slot bookkeeping (integer atomics only -> deterministic output)
// ---------------------------------------------------------------------------
__global__ void count_kernel(const int* __restrict__ topk_id, int* __restrict__ counts) {
    int i = blockIdx.x * 256 + threadIdx.x;
    if (i < T * TOPK) atomicAdd(&counts[topk_id[i]], 1);
}

__global__ void prefix_kernel(const int* __restrict__ counts,
                              int* __restrict__ offs, int* __restrict__ cursor) {
    if (threadIdx.x == 0) {
        int s = 0;
        for (int e = 0; e < E; ++e) { offs[e] = s; cursor[e] = s; s += counts[e]; }
    }
}

__global__ void scatter_kernel(const int* __restrict__ topk_id,
                               const float* __restrict__ topk_w,
                               int* __restrict__ cursor,
                               int* __restrict__ slot_tok,
                               float* __restrict__ slot_w,
                               int* __restrict__ slot_of) {
    int i = blockIdx.x * 256 + threadIdx.x;
    if (i >= T * TOPK) return;
    int e = topk_id[i];
    int slot = atomicAdd(&cursor[e], 1);
    slot_tok[slot] = i >> 2;     // token id
    slot_w[slot] = topk_w[i];
    slot_of[i] = slot;
}

// ---------------------------------------------------------------------------
// WMMA fragment helpers (16x16x32 f16, wave32; A/B k order is this code's contiguous choice).
// A (16x32 f16): lane L -> row M=L&15, hi=L>>4; halves 0..15 = K[k0+16hi .. +15]
//                (contiguous, matching B)
// B (32x16 f16): lane L -> col N=L&15; lanes 0-15 hold K=k0..k0+15,
//                lanes 16-31 hold K=k0+16..k0+31 (contiguous in transposed wt)
// C/D (16x16 f32): lane L -> col N=L&15, vgpr v -> row v + 8*(L>>4)
// ---------------------------------------------------------------------------
__device__ __forceinline__ v16h load_a_frag(const _Float16* arow, int k0, int hi) {
    v8h a0 = *(const v8h*)(arow + k0 + 16 * hi);
    v8h a1 = *(const v8h*)(arow + k0 + 16 * hi + 8);
    return __builtin_shufflevector(a0, a1, 0,1,2,3,4,5,6,7,8,9,10,11,12,13,14,15);
}
__device__ __forceinline__ v16h load_b_frag(const _Float16* brow, int k0, int kb) {
    v8h b0 = *(const v8h*)(brow + k0 + kb);
    v8h b1 = *(const v8h*)(brow + k0 + kb + 8);
    return __builtin_shufflevector(b0, b1, 0,1,2,3,4,5,6,7,8,9,10,11,12,13,14,15);
}
__device__ __forceinline__ v8f wmma16(v16h a, v16h b, v8f c) {
    return __builtin_amdgcn_wmma_f32_16x16x32_f16(false, a, false, b, (short)0, c, false, false);
}
__device__ __forceinline__ float silu_f(float g) { return g / (1.f + expf(-g)); }

// ---------------------------------------------------------------------------
// Fused gate/up GEMM, register-blocked 32M x 32N per wave, K = H.
// 12 b128 loads per 8 WMMAs (1.5 loads/wmma).
// ---------------------------------------------------------------------------
__device__ __forceinline__ void gateup_tile32(const _Float16* arow0, const _Float16* arow1,
                                              const _Float16* bg0, const _Float16* bg1,
                                              const _Float16* bu0, const _Float16* bu1,
                                              int hi, v8f g[2][2], v8f u[2][2]) {
    int kb = hi * 16;
    for (int k0 = 0; k0 < H; k0 += 32) {
        v16h a0 = load_a_frag(arow0, k0, hi);
        v16h a1 = load_a_frag(arow1, k0, hi);
        v16h bgv0 = load_b_frag(bg0, k0, kb);
        v16h bgv1 = load_b_frag(bg1, k0, kb);
        v16h buv0 = load_b_frag(bu0, k0, kb);
        v16h buv1 = load_b_frag(bu1, k0, kb);
        g[0][0] = wmma16(a0, bgv0, g[0][0]);
        g[0][1] = wmma16(a0, bgv1, g[0][1]);
        g[1][0] = wmma16(a1, bgv0, g[1][0]);
        g[1][1] = wmma16(a1, bgv1, g[1][1]);
        u[0][0] = wmma16(a0, buv0, u[0][0]);
        u[0][1] = wmma16(a0, buv1, u[0][1]);
        u[1][0] = wmma16(a1, buv0, u[1][0]);
        u[1][1] = wmma16(a1, buv1, u[1][1]);
    }
}

// Down GEMM, register-blocked 32M x 32N per wave, K = I.
// 8 b128 loads per 4 WMMAs (2.0 loads/wmma).
__device__ __forceinline__ void down_tile32(const _Float16* arow0, const _Float16* arow1,
                                            const _Float16* b0, const _Float16* b1,
                                            int hi, v8f acc[2][2]) {
    int kb = hi * 16;
    for (int k0 = 0; k0 < I; k0 += 32) {
        v16h a0 = load_a_frag(arow0, k0, hi);
        v16h a1 = load_a_frag(arow1, k0, hi);
        v16h bv0 = load_b_frag(b0, k0, kb);
        v16h bv1 = load_b_frag(b1, k0, kb);
        acc[0][0] = wmma16(a0, bv0, acc[0][0]);
        acc[0][1] = wmma16(a0, bv1, acc[0][1]);
        acc[1][0] = wmma16(a1, bv0, acc[1][0]);
        acc[1][1] = wmma16(a1, bv1, acc[1][1]);
    }
}

// ---------------------------------------------------------------------------
// Routed experts gate/up: mid[slot, I] = silu(X@Wg) * (X@Wu).
// block = 256 thr = 8 waves; per-wave tile M=32 (tokens) x N=32 (cols);
// block tile M=32 x N=256. grid = (I/256, T/32, E); early-exit on counts[e].
// ---------------------------------------------------------------------------
__global__ void moe_gateup_kernel(const _Float16* __restrict__ xh,
                                  const _Float16* __restrict__ wg_t,   // [E][I][H]
                                  const _Float16* __restrict__ wu_t,   // [E][I][H]
                                  const int* __restrict__ counts,
                                  const int* __restrict__ offs,
                                  const int* __restrict__ slot_tok,
                                  _Float16* __restrict__ mid) {        // [S][I]
    int e = blockIdx.z;
    int cnt = counts[e];
    int mbase = blockIdx.y * 32;
    if (mbase >= cnt) return;
    int off = offs[e];
    int wave = threadIdx.x >> 5, lane = threadIdx.x & 31;
    int hi = lane >> 4, lo = lane & 15;
    int n0 = blockIdx.x * 256 + wave * 32;    // n-subtiles at n0, n0+16
    int r0 = mbase + lo;      if (r0 >= cnt) r0 = cnt - 1;   // clamp pad rows
    int r1 = mbase + 16 + lo; if (r1 >= cnt) r1 = cnt - 1;
    const _Float16* arow0 = xh + (size_t)slot_tok[off + r0] * H;
    const _Float16* arow1 = xh + (size_t)slot_tok[off + r1] * H;
    const _Float16* bg0 = wg_t + ((size_t)e * I + n0 + lo) * H;
    const _Float16* bu0 = wu_t + ((size_t)e * I + n0 + lo) * H;
    v8f g[2][2] = {}; v8f u[2][2] = {};
    gateup_tile32(arow0, arow1, bg0, bg0 + (size_t)16 * H, bu0, bu0 + (size_t)16 * H,
                  hi, g, u);
    for (int mi = 0; mi < 2; ++mi)
        for (int ni = 0; ni < 2; ++ni)
            for (int v = 0; v < 8; ++v) {
                int row = mbase + 16 * mi + v + 8 * hi;
                if (row < cnt)
                    mid[(size_t)(off + row) * I + n0 + 16 * ni + lo] =
                        (_Float16)(silu_f(g[mi][ni][v]) * u[mi][ni][v]);
            }
}

__global__ void shared_gateup_kernel(const _Float16* __restrict__ xh,
                                     const _Float16* __restrict__ swg_t,  // [I][H]
                                     const _Float16* __restrict__ swu_t,  // [I][H]
                                     _Float16* __restrict__ smid) {       // [T][I]
    int mbase = blockIdx.y * 32;
    int wave = threadIdx.x >> 5, lane = threadIdx.x & 31;
    int hi = lane >> 4, lo = lane & 15;
    int n0 = blockIdx.x * 256 + wave * 32;
    const _Float16* arow0 = xh + (size_t)(mbase + lo) * H;
    const _Float16* arow1 = xh + (size_t)(mbase + 16 + lo) * H;
    const _Float16* bg0 = swg_t + (size_t)(n0 + lo) * H;
    const _Float16* bu0 = swu_t + (size_t)(n0 + lo) * H;
    v8f g[2][2] = {}; v8f u[2][2] = {};
    gateup_tile32(arow0, arow1, bg0, bg0 + (size_t)16 * H, bu0, bu0 + (size_t)16 * H,
                  hi, g, u);
    for (int mi = 0; mi < 2; ++mi)
        for (int ni = 0; ni < 2; ++ni)
            for (int v = 0; v < 8; ++v) {
                int row = mbase + 16 * mi + v + 8 * hi;
                smid[(size_t)row * I + n0 + 16 * ni + lo] =
                    (_Float16)(silu_f(g[mi][ni][v]) * u[mi][ni][v]);
            }
}

// down_out[slot, H] = slot_w[slot] * (mid[slot,:] @ Wd[e]); slots contiguous.
// block tile M=32 x N=256. grid = (H/256, T/32, E).
__global__ void moe_down_kernel(const _Float16* __restrict__ mid,       // [S][I]
                                const _Float16* __restrict__ wd_t,      // [E][H][I]
                                const int* __restrict__ counts,
                                const int* __restrict__ offs,
                                const float* __restrict__ slot_w,
                                float* __restrict__ down_out) {         // [S][H]
    int e = blockIdx.z;
    int cnt = counts[e];
    int mbase = blockIdx.y * 32;
    if (mbase >= cnt) return;
    int off = offs[e];
    int wave = threadIdx.x >> 5, lane = threadIdx.x & 31;
    int hi = lane >> 4, lo = lane & 15;
    int n0 = blockIdx.x * 256 + wave * 32;    // over H
    int r0 = mbase + lo;      if (r0 >= cnt) r0 = cnt - 1;
    int r1 = mbase + 16 + lo; if (r1 >= cnt) r1 = cnt - 1;
    const _Float16* arow0 = mid + (size_t)(off + r0) * I;
    const _Float16* arow1 = mid + (size_t)(off + r1) * I;
    const _Float16* b0 = wd_t + ((size_t)e * H + n0 + lo) * I;
    v8f acc[2][2] = {};
    down_tile32(arow0, arow1, b0, b0 + (size_t)16 * I, hi, acc);
    for (int mi = 0; mi < 2; ++mi)
        for (int ni = 0; ni < 2; ++ni)
            for (int v = 0; v < 8; ++v) {
                int row = mbase + 16 * mi + v + 8 * hi;
                if (row < cnt) {
                    int slot = off + row;
                    down_out[(size_t)slot * H + n0 + 16 * ni + lo] =
                        acc[mi][ni][v] * slot_w[slot];
                }
            }
}

// shared expert down -> writes d_out (initializes every element)
__global__ void shared_down_kernel(const _Float16* __restrict__ smid,   // [T][I]
                                   const _Float16* __restrict__ swd_t,  // [H][I]
                                   float* __restrict__ out) {           // [T][H]
    int mbase = blockIdx.y * 32;
    int wave = threadIdx.x >> 5, lane = threadIdx.x & 31;
    int hi = lane >> 4, lo = lane & 15;
    int n0 = blockIdx.x * 256 + wave * 32;
    const _Float16* arow0 = smid + (size_t)(mbase + lo) * I;
    const _Float16* arow1 = smid + (size_t)(mbase + 16 + lo) * I;
    const _Float16* b0 = swd_t + (size_t)(n0 + lo) * I;
    v8f acc[2][2] = {};
    down_tile32(arow0, arow1, b0, b0 + (size_t)16 * I, hi, acc);
    for (int mi = 0; mi < 2; ++mi)
        for (int ni = 0; ni < 2; ++ni)
            for (int v = 0; v < 8; ++v)
                out[(size_t)(mbase + 16 * mi + v + 8 * hi) * H + n0 + 16 * ni + lo] =
                    acc[mi][ni][v];
}

// out[t,h] += sum_k down_out[slot_of[t,k], h]   (fixed order -> deterministic)
__global__ void combine_kernel(const float* __restrict__ down_out,
                               const int* __restrict__ slot_of,
                               float* __restrict__ out) {
    int idx = blockIdx.x * 256 + threadIdx.x;   // T*H total, H=1024
    int t = idx >> 10;
    int h = idx & (H - 1);
    float acc = out[idx];
    for (int k = 0; k < TOPK; ++k)
        acc += down_out[(size_t)slot_of[t * TOPK + k] * H + h];
    out[idx] = acc;
}

// ---------------------------------------------------------------------------
extern "C" void kernel_launch(void* const* d_in, const int* in_sizes, int n_in,
                              void* d_out, int out_size, void* d_ws, size_t ws_size,
                              hipStream_t stream) {
    const float* x       = (const float*)d_in[0];
    const float* gate_w  = (const float*)d_in[1];
    const float* bias    = (const float*)d_in[2];
    const float* w_gate  = (const float*)d_in[3];
    const float* w_up    = (const float*)d_in[4];
    const float* w_down  = (const float*)d_in[5];
    const float* sw_gate = (const float*)d_in[6];
    const float* sw_up   = (const float*)d_in[7];
    const float* sw_down = (const float*)d_in[8];
    float* out = (float*)d_out;

    // Workspace carve-out (~129 MB), 256B aligned slices
    char* wp = (char*)d_ws;
    auto carve = [&](size_t bytes) {
        char* p = wp; wp += (bytes + 255) & ~(size_t)255; return p;
    };
    _Float16* xh    = (_Float16*)carve((size_t)T * H * 2);
    _Float16* wg_t  = (_Float16*)carve((size_t)E * I * H * 2);
    _Float16* wu_t  = (_Float16*)carve((size_t)E * I * H * 2);
    _Float16* wd_t  = (_Float16*)carve((size_t)E * H * I * 2);
    _Float16* swg_t = (_Float16*)carve((size_t)I * H * 2);
    _Float16* swu_t = (_Float16*)carve((size_t)I * H * 2);
    _Float16* swd_t = (_Float16*)carve((size_t)H * I * 2);
    _Float16* mid   = (_Float16*)carve((size_t)S * I * 2);
    _Float16* smid  = (_Float16*)carve((size_t)T * I * 2);
    float* down_out = (float*)carve((size_t)S * H * 4);
    int*   topk_id  = (int*)carve((size_t)T * TOPK * 4);
    float* topk_w   = (float*)carve((size_t)T * TOPK * 4);
    int*   counts   = (int*)carve(E * 4);
    int*   offs     = (int*)carve(E * 4);
    int*   cursor   = (int*)carve(E * 4);
    int*   slot_tok = (int*)carve(S * 4);
    float* slot_w   = (float*)carve(S * 4);
    int*   slot_of  = (int*)carve((size_t)T * TOPK * 4);

    // 1) activation convert + weight convert/transpose to f16 N-major
    f32_to_f16_kernel<<<(T * H) / 256, 256, 0, stream>>>(x, xh, T * H);
    transpose_f32_f16_kernel<<<dim3(I/32, H/32, E), dim3(32, 8), 0, stream>>>(w_gate, wg_t, H, I);
    transpose_f32_f16_kernel<<<dim3(I/32, H/32, E), dim3(32, 8), 0, stream>>>(w_up,   wu_t, H, I);
    transpose_f32_f16_kernel<<<dim3(H/32, I/32, E), dim3(32, 8), 0, stream>>>(w_down, wd_t, I, H);
    transpose_f32_f16_kernel<<<dim3(I/32, H/32, 1), dim3(32, 8), 0, stream>>>(sw_gate, swg_t, H, I);
    transpose_f32_f16_kernel<<<dim3(I/32, H/32, 1), dim3(32, 8), 0, stream>>>(sw_up,   swu_t, H, I);
    transpose_f32_f16_kernel<<<dim3(H/32, I/32, 1), dim3(32, 8), 0, stream>>>(sw_down, swd_t, I, H);

    // 2) routing (fp32 exact)
    router_kernel<<<T / 8, 256, 0, stream>>>(x, gate_w, bias, topk_id, topk_w);

    // 3) slot bookkeeping
    zero_i32_kernel<<<1, 256, 0, stream>>>(counts, E);
    count_kernel<<<(T * TOPK) / 256, 256, 0, stream>>>(topk_id, counts);
    prefix_kernel<<<1, 32, 0, stream>>>(counts, offs, cursor);
    scatter_kernel<<<(T * TOPK) / 256, 256, 0, stream>>>(topk_id, topk_w, cursor,
                                                         slot_tok, slot_w, slot_of);

    // 4) expert GEMMs (WMMA f16 -> f32, 32x32 register-blocked tiles)
    moe_gateup_kernel<<<dim3(I/256, T/32, E), 256, 0, stream>>>(
        xh, wg_t, wu_t, counts, offs, slot_tok, mid);
    shared_gateup_kernel<<<dim3(I/256, T/32, 1), 256, 0, stream>>>(xh, swg_t, swu_t, smid);
    moe_down_kernel<<<dim3(H/256, T/32, E), 256, 0, stream>>>(
        mid, wd_t, counts, offs, slot_w, down_out);
    shared_down_kernel<<<dim3(H/256, T/32, 1), 256, 0, stream>>>(smid, swd_t, out);

    // 5) combine routed + shared
    combine_kernel<<<(T * H) / 256, 256, 0, stream>>>(down_out, slot_of, out);
}
